// ClusterEncoder_14937896255979
// MI455X (gfx1250) — matrix operand order repaired and checked by reading, not compiled, not run on hardware
//
#include <hip/hip_runtime.h>

#define S_ 4
#define N_ 32768
#define D_ 256
#define K_ 512
#define NH_ 4
#define HD_ 64
#define F_ 1024
#define ROWS_P (S_ * K_)   // 2048 patch rows
#define ROWS_N (S_ * N_)   // 131072 node rows

typedef __attribute__((ext_vector_type(16))) __bf16 v16bf;
typedef __attribute__((ext_vector_type(8)))  __bf16 v8bf;
typedef __attribute__((ext_vector_type(8)))  float  v8f;
typedef __attribute__((vector_size(16)))     int    v4i_b128;   // async builtin's pointee type

#if __has_builtin(__builtin_amdgcn_global_load_async_to_lds_b128) && \
    __has_builtin(__builtin_amdgcn_s_wait_asynccnt)
#define HAVE_ASYNC_LDS 1
#endif

// 16-byte global -> LDS copy; async (ASYNCcnt) on gfx1250 if available.
__device__ __forceinline__ void copy16_g2l(const void* g, void* l) {
#ifdef HAVE_ASYNC_LDS
    __builtin_amdgcn_global_load_async_to_lds_b128(
        (__attribute__((address_space(1))) v4i_b128*)g,
        (__attribute__((address_space(3))) v4i_b128*)l, 0, 0);
#else
    *(v8bf*)l = *(const v8bf*)g;
#endif
}
__device__ __forceinline__ void wait_async_copies() {
#ifdef HAVE_ASYNC_LDS
    __builtin_amdgcn_s_wait_asynccnt(0);
#endif
}

// ---------------------------------------------------------------- utilities
__global__ __launch_bounds__(256) void zero_kernel(float* p, int n) {
    int i = blockIdx.x * 256 + threadIdx.x;
    if (i < n) p[i] = 0.f;
}

// W [Kd][Nd] f32 row-major  ->  Wt [Nd][Kd] bf16 (transposed, for B fragments)
__global__ __launch_bounds__(256) void transpose_bf16_kernel(const float* W, __bf16* Wt,
                                                             int Kd, int Nd) {
    int i = blockIdx.x * 256 + threadIdx.x;
    if (i < Kd * Nd) {
        int k = i / Nd, n = i % Nd;
        Wt[(size_t)n * Kd + k] = (__bf16)W[i];
    }
}

// ------------------------------------------------------- segment-mean pooling
__global__ __launch_bounds__(256) void pool_kernel(const float* H, const int* pids,
                                                   float* sums, float* counts) {
    int row = blockIdx.x;          // 0..ROWS_N-1
    int d   = threadIdx.x;         // 0..255
    int s   = row >> 15;           // row / N_
    int seg = (s << 9) + pids[row];
    atomicAdd(&sums[(size_t)seg * D_ + d], H[(size_t)row * D_ + d]);
    if (d == 0) atomicAdd(&counts[seg], 1.0f);
}

// x = sums/max(cnt,1); mask; y1 = LN1(x) in bf16.  One block per patch row.
__global__ __launch_bounds__(256) void finalize_ln1_kernel(const float* sums, const float* counts,
                                                           const float* g, const float* b,
                                                           float* x, float* mask, __bf16* y) {
    int r = blockIdx.x, t = threadIdx.x;
    float cnt = counts[r];
    float xv  = sums[(size_t)r * D_ + t] / fmaxf(cnt, 1.0f);
    x[(size_t)r * D_ + t] = xv;
    if (t == 0) mask[r] = cnt > 0.f ? 1.f : 0.f;
    __shared__ float s1[256], s2[256];
    s1[t] = xv; s2[t] = xv * xv;
    __syncthreads();
    for (int o = 128; o > 0; o >>= 1) {
        if (t < o) { s1[t] += s1[t + o]; s2[t] += s2[t + o]; }
        __syncthreads();
    }
    float mu  = s1[0] * (1.f / D_);
    float var = s2[0] * (1.f / D_) - mu * mu;
    float rs  = rsqrtf(var + 1e-5f);
    y[(size_t)r * D_ + t] = (__bf16)((xv - mu) * rs * g[t] + b[t]);
}

// plain LN (x2 -> y2 bf16)
__global__ __launch_bounds__(256) void ln_kernel(const float* X, const float* g, const float* b,
                                                 __bf16* Y) {
    int r = blockIdx.x, t = threadIdx.x;
    float xv = X[(size_t)r * D_ + t];
    __shared__ float s1[256], s2[256];
    s1[t] = xv; s2[t] = xv * xv;
    __syncthreads();
    for (int o = 128; o > 0; o >>= 1) {
        if (t < o) { s1[t] += s1[t + o]; s2[t] += s2[t + o]; }
        __syncthreads();
    }
    float mu  = s1[0] * (1.f / D_);
    float var = s2[0] * (1.f / D_) - mu * mu;
    float rs  = rsqrtf(var + 1e-5f);
    Y[(size_t)r * D_ + t] = (__bf16)((xv - mu) * rs * g[t] + b[t]);
}

// --------------------------------------------------- linear-attn small kernels
__global__ __launch_bounds__(256) void kv_kernel(const float* kf, const float* vf,
                                                 float* kv, float* ksum) {
    int b = blockIdx.x;            // s*NH + h
    int s = b / NH_, h = b % NH_;
    int t = threadIdx.x;
    int d  = t >> 2;               // 0..63
    int e0 = (t & 3) * 16;         // 16 e's each
    float acc[16];
#pragma unroll
    for (int i = 0; i < 16; i++) acc[i] = 0.f;
    float ks = 0.f;
    const float* kb = kf + (size_t)s * K_ * D_ + h * HD_;
    const float* vb = vf + (size_t)s * K_ * D_ + h * HD_;
    for (int kk = 0; kk < K_; kk++) {
        float kd = kb[(size_t)kk * D_ + d];
        ks += kd;
        const float* vr = vb + (size_t)kk * D_ + e0;
#pragma unroll
        for (int i = 0; i < 16; i++) acc[i] += kd * vr[i];
    }
    float* kvb = kv + ((size_t)b * HD_ + d) * HD_ + e0;
#pragma unroll
    for (int i = 0; i < 16; i++) kvb[i] = acc[i];
    if (e0 == 0) ksum[(size_t)b * HD_ + d] = ks;
}

__global__ __launch_bounds__(256) void attn_kernel(const float* q, const float* kv,
                                                   const float* ksum, __bf16* attn) {
    int idx = blockIdx.x * 256 + threadIdx.x;
    int row = idx >> 8, col = idx & 255;
    int s = row >> 9, h = col >> 6, e = col & 63;
    const float* qr  = q + (size_t)row * D_ + h * HD_;
    const float* kvp = kv + ((size_t)(s * NH_ + h) * HD_) * HD_ + e;
    const float* ksp = ksum + (size_t)(s * NH_ + h) * HD_;
    float num = 0.f, den = 0.f;
#pragma unroll 8
    for (int d = 0; d < HD_; d++) {
        float qd = qr[d];
        num += qd * kvp[(size_t)d * HD_];
        den += qd * ksp[d];
    }
    attn[idx] = (__bf16)(num / (den + 1e-6f));
}

// --------------------------------------------------------- generic WMMA GEMM
// C[M,Nd] = act( A[M,Kd](bf16) x W + bias ) [+ res]
// Block: 64 rows x 256 cols, BK=32, double-buffered LDS, async global->LDS.
// 8 waves: wave(wm,wn) = 16 rows x 128 cols = 8 accumulators, 8 WMMA / K-step.
// epi: 1=phi  2=phi*mask  3=*mask  4=+res->f32  5=gelu->bf16  6=+res->f32&bf16
// gatherMode: A built on the fly: k<256 from f32 Hg (convert), k>=256 gathered
// bf16 rows of Pg via pidsg (node-MLP concat fusion).
__global__ __launch_bounds__(256) void gemm_bf16_wmma(
    const __bf16* A, const __bf16* Wt, const float* bias,
    float* Cf, __bf16* Cb, const float* res, const float* mask,
    int Kd, int Nd, int epi,
    int gatherMode, const float* Hg, const __bf16* Pg, const int* pidsg) {

    __shared__ __bf16 Asub[2][64][40];    // 64 x 32, padded rows (80B stride)
    __shared__ __bf16 Bsub[2][256][40];   // 256(N) x 32(K) from Wt

    const int gm = blockIdx.y * 64;       // row base
    const int gn = blockIdx.x * 256;      // col base

    const int c  = threadIdx.x;           // cooperative loader id
    const int ar = c >> 2;                // A tile row 0..63
    const int ak = (c & 3) * 8;           // k offset {0,8,16,24}
    const int grow_ld = gm + ar;

    // hoisted gather row (constant across the K loop)
    size_t gatherRow = 0;
    if (gatherMode) {
        int s   = grow_ld >> 15;                 // / N_
        int pid = pidsg[grow_ld];
        gatherRow = (size_t)(s * K_ + pid) * D_; // psu_bf16 row base
    }

    const int wave = threadIdx.x >> 5;
    const int lane = threadIdx.x & 31;
    const int wm   = wave & 3;            // 4 wave rows  -> 16*wm
    const int wn   = wave >> 2;           // 2 wave cols  -> 128*wn
    const int frow = lane & 15;
    const int khiA = (lane >> 4) * 8;     // A interleaved half-group K offset
    const int kbB  = (lane >> 4) * 16;    // unused; B reads use khiA like A

    v8f acc[8];
#pragma unroll
    for (int i = 0; i < 8; i++) acc[i] = (v8f){0.f, 0.f, 0.f, 0.f, 0.f, 0.f, 0.f, 0.f};

    auto load_tiles = [&](int buf, int k0) {
        // A tile: 64x32 = one 16B chunk per thread
        if (gatherMode == 0) {
            copy16_g2l(A + (size_t)grow_ld * Kd + k0 + ak, &Asub[buf][ar][ak]);
        } else if (k0 < 256) {
            const float* hp = Hg + (size_t)grow_ld * D_ + k0 + ak;
#pragma unroll
            for (int i = 0; i < 8; i++) Asub[buf][ar][ak + i] = (__bf16)hp[i];
        } else {
            copy16_g2l(Pg + gatherRow + (k0 - 256) + ak, &Asub[buf][ar][ak]);
        }
        // B tile: 256x32 = four 16B chunks per thread
#pragma unroll
        for (int i = 0; i < 4; i++) {
            int brow = ar + 64 * i;
            copy16_g2l(Wt + (size_t)(gn + brow) * Kd + k0 + ak, &Bsub[buf][brow][ak]);
        }
    };

    load_tiles(0, 0);
    wait_async_copies();
    __syncthreads();

    int cur = 0;
    for (int k0 = 0; k0 < Kd; k0 += 32) {
        int nxt = cur ^ 1;
        if (k0 + 32 < Kd) load_tiles(nxt, k0 + 32);   // prefetch overlaps WMMAs

        // A fragment: row 16*wm+frow, K halves {khiA..+7, 16+khiA..+7}
        const __bf16* arow = &Asub[cur][16 * wm + frow][0];
        v8bf alo = *(const v8bf*)(arow + khiA);
        v8bf ahi = *(const v8bf*)(arow + 16 + khiA);
        v16bf afrag;
#pragma unroll
        for (int i = 0; i < 8; i++) { afrag[i] = alo[i]; afrag[8 + i] = ahi[i]; }

        // Preload ALL 8 B fragments so the ds_loads batch up and the 8 WMMAs
        // can issue back-to-back without per-WMMA s_wait_dscnt serialization.
        v16bf bfrag[8];
#pragma unroll
        for (int cg = 0; cg < 8; cg++) {
            const __bf16* brow = &Bsub[cur][128 * wn + 16 * cg + frow][0];
            v8bf blo = *(const v8bf*)(brow + khiA);
            v8bf bhi = *(const v8bf*)(brow + 16 + khiA);
#pragma unroll
            for (int i = 0; i < 8; i++) { bfrag[cg][i] = blo[i]; bfrag[cg][8 + i] = bhi[i]; }
        }
#pragma unroll
        for (int cg = 0; cg < 8; cg++) {
            acc[cg] = __builtin_amdgcn_wmma_f32_16x16x32_bf16(
                false, afrag, false, bfrag[cg], (short)0, acc[cg], false, false);
        }

        wait_async_copies();   // next tile landed
        __syncthreads();       // all waves done with cur; safe to overwrite
        cur = nxt;
    }

    // ------------------------------- epilogue -------------------------------
    const int colBase = gn + 128 * wn + (lane & 15);
    const int rowHalf = (lane >> 4) * 8;
#pragma unroll
    for (int j = 0; j < 8; j++) {
        int grow = gm + 16 * wm + rowHalf + j;
#pragma unroll
        for (int cg = 0; cg < 8; cg++) {
            int gc  = colBase + 16 * cg;
            float v = acc[cg][j] + bias[gc];
            size_t o = (size_t)grow * Nd + gc;
            switch (epi) {
                case 1:  // phi = elu + 1
                    v = v > 0.f ? v + 1.f : __expf(v);
                    Cf[o] = v; break;
                case 2:  // phi * mask
                    v = v > 0.f ? v + 1.f : __expf(v);
                    Cf[o] = v * mask[grow]; break;
                case 3:  // * mask
                    Cf[o] = v * mask[grow]; break;
                case 4:  // + residual -> f32
                    Cf[o] = v + res[o]; break;
                case 5:  // exact gelu -> bf16
                    v = 0.5f * v * (1.f + erff(v * 0.70710678118654752f));
                    Cb[o] = (__bf16)v; break;
                case 6:  // + residual -> f32 and bf16
                    v += res[o];
                    Cf[o] = v; Cb[o] = (__bf16)v; break;
                default:
                    Cf[o] = v; break;
            }
        }
    }
}

// ------------------------------------------------------------------- launch
extern "C" void kernel_launch(void* const* d_in, const int* in_sizes, int n_in,
                              void* d_out, int out_size, void* d_ws, size_t ws_size,
                              hipStream_t stream) {
    (void)in_sizes; (void)n_in; (void)out_size; (void)ws_size;
    const float* Hn   = (const float*)d_in[0];
    const int*   pids = (const int*)d_in[1];
    const float* ln1g = (const float*)d_in[2];
    const float* ln1b = (const float*)d_in[3];
    const float* Wq = (const float*)d_in[4];  const float* bq = (const float*)d_in[5];
    const float* Wk = (const float*)d_in[6];  const float* bk = (const float*)d_in[7];
    const float* Wv = (const float*)d_in[8];  const float* bv = (const float*)d_in[9];
    const float* Wo = (const float*)d_in[10]; const float* bo = (const float*)d_in[11];
    const float* ln2g = (const float*)d_in[12]; const float* ln2b = (const float*)d_in[13];
    const float* W1 = (const float*)d_in[14]; const float* b1 = (const float*)d_in[15];
    const float* W2 = (const float*)d_in[16]; const float* b2 = (const float*)d_in[17];
    const float* M1 = (const float*)d_in[18]; const float* mb1 = (const float*)d_in[19];
    const float* M2 = (const float*)d_in[20]; const float* mb2 = (const float*)d_in[21];

    float* outH   = (float*)d_out;                               // [S,N,D]
    float* outPsu = (float*)d_out + (size_t)ROWS_N * D_;         // [S,K,D]

    // workspace carve-out (256B aligned)
    char* base = (char*)d_ws;
    size_t off = 0;
    auto alloc = [&](size_t bytes) -> void* {
        void* p = base + off;
        off += (bytes + 255) & ~(size_t)255;
        return p;
    };
    float* sums   = (float*)alloc((size_t)ROWS_P * D_ * 4);
    float* counts = (float*)alloc((size_t)ROWS_P * 4);
    float* x      = (float*)alloc((size_t)ROWS_P * D_ * 4);
    float* maskv  = (float*)alloc((size_t)ROWS_P * 4);
    float* x2     = (float*)alloc((size_t)ROWS_P * D_ * 4);
    float* qf     = (float*)alloc((size_t)ROWS_P * D_ * 4);
    float* kf     = (float*)alloc((size_t)ROWS_P * D_ * 4);
    float* vf     = (float*)alloc((size_t)ROWS_P * D_ * 4);
    float* kvb    = (float*)alloc((size_t)S_ * NH_ * HD_ * HD_ * 4);
    float* ksumb  = (float*)alloc((size_t)S_ * NH_ * HD_ * 4);
    __bf16* y1    = (__bf16*)alloc((size_t)ROWS_P * D_ * 2);
    __bf16* attnb = (__bf16*)alloc((size_t)ROWS_P * D_ * 2);
    __bf16* y2    = (__bf16*)alloc((size_t)ROWS_P * D_ * 2);
    __bf16* hbuf  = (__bf16*)alloc((size_t)ROWS_P * F_ * 2);
    __bf16* psub  = (__bf16*)alloc((size_t)ROWS_P * D_ * 2);
    __bf16* ubuf  = (__bf16*)alloc((size_t)ROWS_N * D_ * 2);
    __bf16* WqT = (__bf16*)alloc((size_t)D_ * D_ * 2);
    __bf16* WkT = (__bf16*)alloc((size_t)D_ * D_ * 2);
    __bf16* WvT = (__bf16*)alloc((size_t)D_ * D_ * 2);
    __bf16* WoT = (__bf16*)alloc((size_t)D_ * D_ * 2);
    __bf16* W1T = (__bf16*)alloc((size_t)D_ * F_ * 2);
    __bf16* W2T = (__bf16*)alloc((size_t)F_ * D_ * 2);
    __bf16* M1T = (__bf16*)alloc((size_t)2 * D_ * D_ * 2);
    __bf16* M2T = (__bf16*)alloc((size_t)D_ * D_ * 2);

    // 1) pooling
    zero_kernel<<<(ROWS_P * D_ + 255) / 256, 256, 0, stream>>>(sums, ROWS_P * D_);
    zero_kernel<<<(ROWS_P + 255) / 256, 256, 0, stream>>>(counts, ROWS_P);
    pool_kernel<<<ROWS_N, 256, 0, stream>>>(Hn, pids, sums, counts);

    // 2) bf16 transposed weights (tiny, once per launch)
    auto tr = [&](const float* W, __bf16* Wt, int Kd, int Nd) {
        transpose_bf16_kernel<<<(Kd * Nd + 255) / 256, 256, 0, stream>>>(W, Wt, Kd, Nd);
    };
    tr(Wq, WqT, D_, D_);  tr(Wk, WkT, D_, D_);
    tr(Wv, WvT, D_, D_);  tr(Wo, WoT, D_, D_);
    tr(W1, W1T, D_, F_);  tr(W2, W2T, F_, D_);
    tr(M1, M1T, 2 * D_, D_); tr(M2, M2T, D_, D_);

    // 3) x, mask, LN1
    finalize_ln1_kernel<<<ROWS_P, 256, 0, stream>>>(sums, counts, ln1g, ln1b, x, maskv, y1);

    // 4) Q/K/V projections (WMMA), phi / mask epilogues
    dim3 gP(D_ / 256, ROWS_P / 64);              // (1, 32)
    gemm_bf16_wmma<<<gP, 256, 0, stream>>>(y1, WqT, bq, qf, nullptr, nullptr, maskv,
                                           D_, D_, 1, 0, nullptr, nullptr, nullptr);
    gemm_bf16_wmma<<<gP, 256, 0, stream>>>(y1, WkT, bk, kf, nullptr, nullptr, maskv,
                                           D_, D_, 2, 0, nullptr, nullptr, nullptr);
    gemm_bf16_wmma<<<gP, 256, 0, stream>>>(y1, WvT, bv, vf, nullptr, nullptr, maskv,
                                           D_, D_, 3, 0, nullptr, nullptr, nullptr);

    // 5) linear attention: kv/ksum, then num/den combine -> bf16
    kv_kernel<<<S_ * NH_, 256, 0, stream>>>(kf, vf, kvb, ksumb);
    attn_kernel<<<ROWS_P * D_ / 256, 256, 0, stream>>>(qf, kvb, ksumb, attnb);

    // 6) O projection + residual(x) -> x2
    gemm_bf16_wmma<<<gP, 256, 0, stream>>>(attnb, WoT, bo, x2, nullptr, x, nullptr,
                                           D_, D_, 4, 0, nullptr, nullptr, nullptr);

    // 7) LN2, FFN
    ln_kernel<<<ROWS_P, 256, 0, stream>>>(x2, ln2g, ln2b, y2);
    dim3 gF1(F_ / 256, ROWS_P / 64);             // (4, 32)
    gemm_bf16_wmma<<<gF1, 256, 0, stream>>>(y2, W1T, b1, nullptr, hbuf, nullptr, nullptr,
                                            D_, F_, 5, 0, nullptr, nullptr, nullptr);
    gemm_bf16_wmma<<<gP, 256, 0, stream>>>(hbuf, W2T, b2, outPsu, psub, x2, nullptr,
                                           F_, D_, 6, 0, nullptr, nullptr, nullptr);

    // 8) node MLP with fused gather+concat A-tile (A streamed exactly once)
    dim3 gM(D_ / 256, ROWS_N / 64);              // (1, 2048)
    gemm_bf16_wmma<<<gM, 256, 0, stream>>>(nullptr, M1T, mb1, nullptr, ubuf, nullptr, nullptr,
                                           2 * D_, D_, 5, 1, Hn, psub, pids);
    gemm_bf16_wmma<<<gM, 256, 0, stream>>>(ubuf, M2T, mb2, outH, nullptr, Hn, nullptr,
                                           D_, D_, 4, 0, nullptr, nullptr, nullptr);
}
